// BiMambaBlock_44392781971761
// MI455X (gfx1250) — matrix operand order repaired and checked by reading, not compiled, not run on hardware
//
#include <hip/hip_runtime.h>
#include <hip/hip_bf16.h>
#include <math.h>

// ---------------- shapes ----------------
#define BATCH   8
#define SEQ     1024
#define DMODEL  512
#define DINNER  1024
#define DSTATE  16
#define DTRANK  32
#define NTOK    (BATCH * SEQ)          // 8192
#define LN_EPS  1e-5f

// ---------------- WMMA types ----------------
typedef __bf16 bf16_t;
typedef bf16_t v16bf __attribute__((ext_vector_type(16)));
typedef float  v8f   __attribute__((ext_vector_type(8)));

union BFrag { v16bf v; uint4 q[2]; };

// ---------------- helpers ----------------
__device__ __forceinline__ unsigned short f2bf(float f) {
    unsigned int u = __float_as_uint(f);
    unsigned int r = (u + 0x7FFFu + ((u >> 16) & 1u)) >> 16;
    return (unsigned short)r;
}
__device__ __forceinline__ float silu(float x) {
    return x / (1.0f + __expf(-x));
}
// generic shared pointer -> 32-bit LDS byte offset (low 32 bits per aperture rules)
__device__ __forceinline__ unsigned lds_off(const void* p) {
    return (unsigned)(unsigned long long)(uintptr_t)p;
}
// one lane moves 16B global -> LDS via the CDNA5 async path (ASYNCcnt)
__device__ __forceinline__ void async_b128(unsigned ldsaddr, const void* gaddr) {
    asm volatile("global_load_async_to_lds_b128 %0, %1, off"
                 :: "v"(ldsaddr), "v"((unsigned long long)(uintptr_t)gaddr)
                 : "memory");
}
__device__ __forceinline__ void wait_async0() {
    asm volatile("s_wait_asynccnt 0x0" ::: "memory");
}

// =====================================================================
// GEMM: C[M,N] (f32) = A[M,K](bf16, row-major) * B[N,K](bf16, row-major)^T
// Block tile 128x64x32, 256 threads = 8 waves (4 along M, 2 along N).
// Each wave owns a 32x32 output tile -> 4 accumulators, 4 WMMA / K-step.
// Staging: double-buffered GLOBAL_LOAD_ASYNC_TO_LDS_B128, one barrier/step.
// =====================================================================
#define TILE_M 128
#define TILE_N 64
#define TILE_K 32
#define LDSS   40   // 32 K-elems + 8 pad (80B rows, 16B aligned)

__global__ __launch_bounds__(256)
void gemm_bf16_wmma(const unsigned short* __restrict__ A,
                    const unsigned short* __restrict__ B,
                    float* __restrict__ C,
                    int M, int N, int K)
{
    __shared__ __align__(16) unsigned short lA[2][TILE_M * LDSS];
    __shared__ __align__(16) unsigned short lB[2][TILE_N * LDSS];

    const int tid  = threadIdx.x;
    const int lane = tid & 31;
    const int wave = tid >> 5;
    const int wm   = wave & 3;       // 0..3 -> 32-row sub-tile
    const int wn   = wave >> 2;      // 0..1 -> 32-col sub-tile
    const int half = lane >> 4;      // 0 or 1
    const int sub  = lane & 15;

    const int m0 = blockIdx.x * TILE_M;
    const int n0 = blockIdx.y * TILE_N;

    // cooperative async-load pattern (16B per lane per issue):
    //   A tile 128x32: two issues/thread, id = tid + 256*j, row=id>>2, col=(id&3)*8
    //   B tile  64x32: one issue/thread,  row=tid>>2,       col=(tid&3)*8
    const int ar0 = (tid)       >> 2, ac0 = ((tid)       & 3) * 8;
    const int ar1 = (tid + 256) >> 2, ac1 = ((tid + 256) & 3) * 8;
    const int br  = tid >> 2,         bc  = (tid & 3) * 8;

    unsigned ldsA0[2], ldsA1[2], ldsB[2];
    #pragma unroll
    for (int bfi = 0; bfi < 2; ++bfi) {
        ldsA0[bfi] = lds_off(&lA[bfi][ar0 * LDSS + ac0]);
        ldsA1[bfi] = lds_off(&lA[bfi][ar1 * LDSS + ac1]);
        ldsB[bfi]  = lds_off(&lB[bfi][br  * LDSS + bc]);
    }
    const unsigned short* gA0 = A + (size_t)(m0 + ar0) * K + ac0;
    const unsigned short* gA1 = A + (size_t)(m0 + ar1) * K + ac1;
    const unsigned short* gB  = B + (size_t)(n0 + br)  * K + bc;

    v8f acc00 = {}, acc01 = {}, acc10 = {}, acc11 = {};

    const int nk = K / TILE_K;
    // prologue: stage tile 0 into buffer 0
    async_b128(ldsA0[0], gA0);
    async_b128(ldsA1[0], gA1);
    async_b128(ldsB[0],  gB);

    for (int kt = 0; kt < nk; ++kt) {
        const int cur = kt & 1, nxt = cur ^ 1;
        wait_async0();          // this wave's pieces of tile kt have landed
        __syncthreads();        // everyone landed; prior reads of 'nxt' done

        if (kt + 1 < nk) {      // stage tile kt+1 into the other buffer
            const size_t adv = (size_t)(kt + 1) * TILE_K;
            async_b128(ldsA0[nxt], gA0 + adv);
            async_b128(ldsA1[nxt], gA1 + adv);
            async_b128(ldsB[nxt],  gB  + adv);
            __builtin_prefetch(gA0 + adv + TILE_K, 0, 1);  // L2 warm for kt+2
        }

        // fragments for a 32x32 wave tile
        BFrag fa0, fa1, fb0, fb1;
        {
            const unsigned short* a0 = &lA[cur][(wm * 32 + sub)      * LDSS + 8 * half];
            const unsigned short* a1 = &lA[cur][(wm * 32 + 16 + sub) * LDSS + 8 * half];
            fa0.q[0] = *(const uint4*)(a0);
            fa0.q[1] = *(const uint4*)(a0 + 16);
            fa1.q[0] = *(const uint4*)(a1);
            fa1.q[1] = *(const uint4*)(a1 + 16);
            const unsigned short* b0 = &lB[cur][(wn * 32 + sub)      * LDSS + 8 * half];
            const unsigned short* b1 = &lB[cur][(wn * 32 + 16 + sub) * LDSS + 8 * half];
            fb0.q[0] = *(const uint4*)(b0);
            fb0.q[1] = *(const uint4*)(b0 + 16);
            fb1.q[0] = *(const uint4*)(b1);
            fb1.q[1] = *(const uint4*)(b1 + 16);
        }
        acc00 = __builtin_amdgcn_wmma_f32_16x16x32_bf16(false, fa0.v, false, fb0.v,
                                                        (short)0, acc00, false, false);
        acc01 = __builtin_amdgcn_wmma_f32_16x16x32_bf16(false, fa0.v, false, fb1.v,
                                                        (short)0, acc01, false, false);
        acc10 = __builtin_amdgcn_wmma_f32_16x16x32_bf16(false, fa1.v, false, fb0.v,
                                                        (short)0, acc10, false, false);
        acc11 = __builtin_amdgcn_wmma_f32_16x16x32_bf16(false, fa1.v, false, fb1.v,
                                                        (short)0, acc11, false, false);
    }

    // C/D layout: VGPR r -> (M = 8*half + r, N = sub)
    #pragma unroll
    for (int r = 0; r < 8; ++r) {
        const int mrow0 = m0 + wm * 32 + 8 * half + r;
        const int mrow1 = mrow0 + 16;
        const int ncol0 = n0 + wn * 32 + sub;
        C[(size_t)mrow0 * N + ncol0]      = acc00[r];
        C[(size_t)mrow0 * N + ncol0 + 16] = acc01[r];
        C[(size_t)mrow1 * N + ncol0]      = acc10[r];
        C[(size_t)mrow1 * N + ncol0 + 16] = acc11[r];
    }
}

// =====================================================================
// elementwise / prep kernels
// =====================================================================
__global__ void cast_bf16_kernel(const float* __restrict__ in,
                                 unsigned short* __restrict__ out, int n) {
    int i = blockIdx.x * blockDim.x + threadIdx.x;
    if (i < n) out[i] = f2bf(in[i]);
}

// x (B,L,D) -> bf16, optionally flipped along L
__global__ void prep_x_kernel(const float* __restrict__ x,
                              unsigned short* __restrict__ out, int flip) {
    int i = blockIdx.x * blockDim.x + threadIdx.x;
    if (i >= NTOK * DMODEL) return;
    int d = i & (DMODEL - 1);
    int l = (i >> 9) & (SEQ - 1);
    int b = i >> 19;
    int ls = flip ? (SEQ - 1 - l) : l;
    out[i] = f2bf(x[((size_t)b << 19) | ((size_t)ls << 9) | d]);
}

// causal depthwise conv (width 4) + bias + SiLU; writes f32 + bf16 copies
__global__ void conv_silu_kernel(const float* __restrict__ xz,
                                 const float* __restrict__ conv_w,
                                 const float* __restrict__ conv_b,
                                 float* __restrict__ xc,
                                 unsigned short* __restrict__ xcbf) {
    int i = blockIdx.x * blockDim.x + threadIdx.x;
    if (i >= NTOK * DINNER) return;
    int e   = i & (DINNER - 1);
    int row = i >> 10;
    int l   = row & (SEQ - 1);
    int rb  = row - l;                 // batch base row
    float s = conv_b[e];
    #pragma unroll
    for (int j = 0; j < 4; ++j) {
        int lj = l - 3 + j;
        if (lj >= 0)
            s += conv_w[e * 4 + j] * xz[(size_t)(rb + lj) * (2 * DINNER) + e];
    }
    float v = silu(s);
    xc[i]   = v;
    xcbf[i] = f2bf(v);
}

// cast the dt_rank slice of dbl to bf16 for the dt GEMM
__global__ void dt_cast_kernel(const float* __restrict__ dbl,
                               unsigned short* __restrict__ dtbf) {
    int i = blockIdx.x * blockDim.x + threadIdx.x;
    if (i >= NTOK * DTRANK) return;
    int r = i & (DTRANK - 1);
    int row = i >> 5;
    dtbf[i] = f2bf(dbl[(size_t)row * (DTRANK + 2 * DSTATE) + r]);
}

// delta = softplus(dtt + dt_b), in place
__global__ void softplus_kernel(float* __restrict__ dtt,
                                const float* __restrict__ dt_b) {
    int i = blockIdx.x * blockDim.x + threadIdx.x;
    if (i >= NTOK * DINNER) return;
    float v = dtt[i] + dt_b[i & (DINNER - 1)];
    dtt[i] = (v > 20.0f) ? v : log1pf(__expf(v));
}

// =====================================================================
// selective scan: one wave handles 2 channels x 16 states.
// lane = (sub-channel)*16 + state.  Sequential over L; shfl reduction
// across the 16 states; lanes 0/16 emit gated output (bf16).
// =====================================================================
__global__ __launch_bounds__(256)
void scan_kernel(const float* __restrict__ delta,   // [NTOK][DINNER]
                 const float* __restrict__ xc,      // [NTOK][DINNER]
                 const float* __restrict__ dbl,     // [NTOK][64] (B:32..47 C:48..63)
                 const float* __restrict__ xz,      // [NTOK][2*DINNER] (z in hi half)
                 const float* __restrict__ A_log,   // [DINNER][16]
                 const float* __restrict__ Dv,      // [DINNER]
                 unsigned short* __restrict__ ybf)  // [NTOK][DINNER]
{
    const int lane  = threadIdx.x & 31;
    const int widx  = blockIdx.x * 8 + (threadIdx.x >> 5);  // global wave id
    const int b     = widx >> 9;                            // / (DINNER/2)
    const int epair = widx & 511;
    const int e     = epair * 2 + (lane >> 4);
    const int n     = lane & 15;

    const float Aval = -__expf(A_log[e * DSTATE + n]);
    const float De   = Dv[e];

    float h = 0.0f;
    for (int l = 0; l < SEQ; ++l) {
        const int row = (b << 10) + l;
        const float d  = delta[(size_t)row * DINNER + e];
        const float xv = xc[(size_t)row * DINNER + e];
        const float bv = dbl[(size_t)row * 64 + 32 + n];
        const float cv = dbl[(size_t)row * 64 + 48 + n];
        h = __expf(d * Aval) * h + d * bv * xv;
        float p = h * cv;
        p += __shfl_xor(p, 1);
        p += __shfl_xor(p, 2);
        p += __shfl_xor(p, 4);
        p += __shfl_xor(p, 8);
        if (n == 0) {
            float z = xz[(size_t)row * (2 * DINNER) + DINNER + e];
            float y = (p + xv * De) * silu(z);
            ybf[(size_t)row * DINNER + e] = f2bf(y);
        }
    }
}

// =====================================================================
// final: out = LN( 0.5*(fwd + flip(bwd)) + x ) ; one block per token
// =====================================================================
__global__ __launch_bounds__(256)
void final_ln_kernel(const float* __restrict__ outf,
                     const float* __restrict__ outb,   // in flipped frame
                     const float* __restrict__ x,
                     const float* __restrict__ g,
                     const float* __restrict__ bb,
                     float* __restrict__ out)
{
    __shared__ float red[256];
    const int tid  = threadIdx.x;
    const int row  = blockIdx.x;
    const int b    = row >> 10;
    const int l    = row & (SEQ - 1);
    const int frow = (b << 10) | (SEQ - 1 - l);

    const size_t base  = (size_t)row  * DMODEL;
    const size_t fbase = (size_t)frow * DMODEL;

    float v0 = 0.5f * (outf[base + tid]        + outb[fbase + tid])        + x[base + tid];
    float v1 = 0.5f * (outf[base + tid + 256]  + outb[fbase + tid + 256])  + x[base + tid + 256];

    red[tid] = v0 + v1;
    __syncthreads();
    for (int s = 128; s > 0; s >>= 1) {
        if (tid < s) red[tid] += red[tid + s];
        __syncthreads();
    }
    const float mu = red[0] * (1.0f / DMODEL);
    __syncthreads();

    float d0 = v0 - mu, d1 = v1 - mu;
    red[tid] = d0 * d0 + d1 * d1;
    __syncthreads();
    for (int s = 128; s > 0; s >>= 1) {
        if (tid < s) red[tid] += red[tid + s];
        __syncthreads();
    }
    const float rstd = rsqrtf(red[0] * (1.0f / DMODEL) + LN_EPS);
    __syncthreads();

    out[base + tid]       = d0 * rstd * g[tid]       + bb[tid];
    out[base + tid + 256] = d1 * rstd * g[tid + 256] + bb[tid + 256];
}

// =====================================================================
// host side
// =====================================================================
extern "C" void kernel_launch(void* const* d_in, const int* in_sizes, int n_in,
                              void* d_out, int out_size, void* d_ws, size_t ws_size,
                              hipStream_t stream)
{
    (void)in_sizes; (void)n_in; (void)out_size; (void)ws_size;
    const float* x    = (const float*)d_in[0];
    const float* ln_g = (const float*)d_in[19];
    const float* ln_b = (const float*)d_in[20];
    float* out = (float*)d_out;

    // workspace carve-out (256B aligned)
    char* ws = (char*)d_ws;
    size_t off = 0;
    auto take = [&](size_t bytes) -> char* {
        char* p = ws + off;
        off += (bytes + 255) & ~(size_t)255;
        return p;
    };
    unsigned short* xbf   = (unsigned short*)take((size_t)NTOK * DMODEL * 2);
    unsigned short* w_in  = (unsigned short*)take((size_t)2 * DINNER * DMODEL * 2);
    unsigned short* w_xp  = (unsigned short*)take((size_t)64 * DINNER * 2);
    unsigned short* w_dt  = (unsigned short*)take((size_t)DINNER * DTRANK * 2);
    unsigned short* w_out = (unsigned short*)take((size_t)DMODEL * DINNER * 2);
    float*          xz    = (float*)take((size_t)NTOK * 2 * DINNER * 4);
    float*          xc    = (float*)take((size_t)NTOK * DINNER * 4);
    unsigned short* xcbf  = (unsigned short*)take((size_t)NTOK * DINNER * 2);
    float*          dbl   = (float*)take((size_t)NTOK * 64 * 4);
    unsigned short* dtbf  = (unsigned short*)take((size_t)NTOK * DTRANK * 2);
    float*          dtt   = (float*)take((size_t)NTOK * DINNER * 4);   // -> delta
    unsigned short* ybf   = (unsigned short*)take((size_t)NTOK * DINNER * 2);
    float*          outF  = (float*)take((size_t)NTOK * DMODEL * 4);
    float*          outB  = (float*)take((size_t)NTOK * DMODEL * 4);

    auto cgrid = [](int n) { return dim3((n + 255) / 256); };

    for (int dir = 0; dir < 2; ++dir) {
        const int base = 1 + dir * 9;
        const float* in_proj  = (const float*)d_in[base + 0];
        const float* conv_w   = (const float*)d_in[base + 1];
        const float* conv_b   = (const float*)d_in[base + 2];
        const float* x_proj   = (const float*)d_in[base + 3];
        const float* dt_w     = (const float*)d_in[base + 4];
        const float* dt_b     = (const float*)d_in[base + 5];
        const float* A_log    = (const float*)d_in[base + 6];
        const float* Dvp      = (const float*)d_in[base + 7];
        const float* out_proj = (const float*)d_in[base + 8];
        float* outD = dir ? outB : outF;

        // casts
        prep_x_kernel<<<cgrid(NTOK * DMODEL), 256, 0, stream>>>(x, xbf, dir);
        cast_bf16_kernel<<<cgrid(2 * DINNER * DMODEL), 256, 0, stream>>>(in_proj,  w_in,  2 * DINNER * DMODEL);
        cast_bf16_kernel<<<cgrid(64 * DINNER),          256, 0, stream>>>(x_proj,   w_xp,  64 * DINNER);
        cast_bf16_kernel<<<cgrid(DINNER * DTRANK),      256, 0, stream>>>(dt_w,     w_dt,  DINNER * DTRANK);
        cast_bf16_kernel<<<cgrid(DMODEL * DINNER),      256, 0, stream>>>(out_proj, w_out, DMODEL * DINNER);

        // xz = x @ in_proj^T   (8192 x 2048 x 512)
        gemm_bf16_wmma<<<dim3(NTOK / TILE_M, (2 * DINNER) / TILE_N), 256, 0, stream>>>(
            xbf, w_in, xz, NTOK, 2 * DINNER, DMODEL);

        // causal conv + SiLU
        conv_silu_kernel<<<cgrid(NTOK * DINNER), 256, 0, stream>>>(xz, conv_w, conv_b, xc, xcbf);

        // dbl = xc @ x_proj^T   (8192 x 64 x 1024)
        gemm_bf16_wmma<<<dim3(NTOK / TILE_M, 64 / TILE_N), 256, 0, stream>>>(
            xcbf, w_xp, dbl, NTOK, 64, DINNER);

        dt_cast_kernel<<<cgrid(NTOK * DTRANK), 256, 0, stream>>>(dbl, dtbf);

        // dtt = dt @ dt_w^T   (8192 x 1024 x 32)
        gemm_bf16_wmma<<<dim3(NTOK / TILE_M, DINNER / TILE_N), 256, 0, stream>>>(
            dtbf, w_dt, dtt, NTOK, DINNER, DTRANK);

        softplus_kernel<<<cgrid(NTOK * DINNER), 256, 0, stream>>>(dtt, dt_b);

        // selective scan + skip + gate  (4096 waves)
        scan_kernel<<<dim3(512), 256, 0, stream>>>(dtt, xc, dbl, xz, A_log, Dvp, ybf);

        // outD = y @ out_proj^T   (8192 x 512 x 1024)
        gemm_bf16_wmma<<<dim3(NTOK / TILE_M, DMODEL / TILE_N), 256, 0, stream>>>(
            ybf, w_out, outD, NTOK, DMODEL, DINNER);
    }

    // combine + residual + LayerNorm
    final_ln_kernel<<<dim3(NTOK), 256, 0, stream>>>(outF, outB, x, ln_g, ln_b, out);
}
